// Attention_46540265620139
// MI455X (gfx1250) — matrix operand order repaired and checked by reading, not compiled, not run on hardware
//
#include <hip/hip_runtime.h>
#include <hip/hip_bf16.h>
#include <math.h>

#define CONV_DIM 2048
#define LSTM_DIM 512
#define ATT_DIM  512
#define BATCH    256
#define NPIX     196
#define NROWS    (BATCH * NPIX)   // 50176 = 16 * 3136
#define KSTEPS   (CONV_DIM / 32)  // 64

typedef __attribute__((ext_vector_type(16))) __bf16 v16bf;
typedef __attribute__((ext_vector_type(8)))  float  v8f;
typedef __attribute__((ext_vector_type(4)))  float  f32x4;
typedef __attribute__((ext_vector_type(2)))  float  f32x2;

union FragBF { v16bf v; unsigned int u[8]; };

// round-to-nearest-even f32->bf16, packed pair (lo = a, hi = b)
__device__ __forceinline__ unsigned int pack_bf16(float a, float b) {
    union { float f; unsigned int i; } ua, ub;
    ua.f = a; ub.f = b;
    unsigned int ha = (ua.i + 0x7FFFu + ((ua.i >> 16) & 1u)) >> 16;
    unsigned int hb = (ub.i + 0x7FFFu + ((ub.i >> 16) & 1u)) >> 16;
    return (hb << 16) | (ha & 0xFFFFu);
}

// ---------------------------------------------------------------------------
// Pass 0: Wc (f32, [A=512][C=2048] row-major) -> packed bf16 pairs in ws
// ---------------------------------------------------------------------------
__global__ __launch_bounds__(256) void wc_to_bf16_kernel(
    const float* __restrict__ Wc, unsigned int* __restrict__ WcB) {
    int idx = blockIdx.x * 256 + threadIdx.x;          // pair index
    const f32x2* W2 = (const f32x2*)Wc;
    f32x2 v = W2[idx];
    WcB[idx] = pack_bf16(v[0], v[1]);
}

// ---------------------------------------------------------------------------
// Pass 1: u[b][a] = bc[a] + bl[a] + dot(lstm[b], Wl[a])    (tiny GEMM)
// ---------------------------------------------------------------------------
__global__ __launch_bounds__(256) void att2_kernel(
    const float* __restrict__ lstm, const float* __restrict__ Wl,
    const float* __restrict__ bc,   const float* __restrict__ bl,
    float* __restrict__ u) {
    __shared__ float h[LSTM_DIM];
    int b = blockIdx.x, t = threadIdx.x;
    for (int k = t; k < LSTM_DIM; k += 256) h[k] = lstm[b * LSTM_DIM + k];
    __syncthreads();
    for (int a = t; a < ATT_DIM; a += 256) {
        float acc = bc[a] + bl[a];
        const float* w = Wl + (size_t)a * LSTM_DIM;
        for (int k = 0; k < LSTM_DIM; ++k) acc += h[k] * w[k];
        u[b * ATT_DIM + a] = acc;
    }
}

// ---------------------------------------------------------------------------
// Pass 2: fused  att[r] = Wf . relu( conv[r,:] @ Wc^T + u[b(r),:] ) + bf
// One block = 16 rows (M tile), full N=512 across 8 waves (4 x 16-N each),
// K loop of 32 using v_wmma_f32_16x16x32_bf16. Double-buffered LDS A tile
// (one barrier per K-step). att1 never hits memory.
// ---------------------------------------------------------------------------
__global__ __launch_bounds__(256) void att_logits_kernel(
    const float*        __restrict__ conv,   // (NROWS, CONV_DIM) f32
    const unsigned int* __restrict__ WcB,    // (ATT_DIM, CONV_DIM/2) bf16 pairs
    const float*        __restrict__ u,      // (BATCH, ATT_DIM)
    const float*        __restrict__ Wf,     // (ATT_DIM)
    const float*        __restrict__ bfp,    // (1)
    float*              __restrict__ att)    // (NROWS)
{
    __shared__ unsigned int AsU[2][16 * 16]; // 2 x (16 rows x 16 bf16-pairs)
    __shared__ float attp[16];
    __shared__ int   bidx[16];               // per-row batch index

    const int t    = threadIdx.x;
    const int lane = t & 31;
    const int wave = t >> 5;                 // 0..7
    const int half = lane >> 4;              // 0/1
    const int l16  = lane & 15;
    const int rowbase = blockIdx.x * 16;

    if (t < 16) {
        attp[t] = 0.0f;
        bidx[t] = (rowbase + t) / NPIX;      // 16 divides per block, not per lane
    }

    v8f acc[4];
    for (int c = 0; c < 4; ++c)
        for (int i = 0; i < 8; ++i) acc[c][i] = 0.0f;

    // cooperative A-tile load mapping: thread t -> (row, K-pair), coalesced in K
    const int arow  = t >> 4;                // 0..15
    const int apair = t & 15;                // pair p -> K = 2p, 2p+1
    const f32x2* convRow =
        (const f32x2*)(conv + (size_t)(rowbase + arow) * CONV_DIM);

    // preload K-tile 0 into buffer 0 (non-temporal: conv stream has no reuse)
    {
        f32x2 av = __builtin_nontemporal_load(convRow + apair);
        AsU[0][arow * 16 + apair] = pack_bf16(av[0], av[1]);
    }
    __syncthreads();

    for (int it = 0; it < KSTEPS; ++it) {
        const int cur = it & 1;

        // prefetch next K-tile into the other buffer (hidden under WMMAs)
        if (it + 1 < KSTEPS) {
            f32x2 av = __builtin_nontemporal_load(
                convRow + ((it + 1) * 16 + apair));
            AsU[cur ^ 1][arow * 16 + apair] = pack_bf16(av[0], av[1]);
        }

        // A fragment (ISA 16-bit A 16x32): VGPR j holds K = (j<4 ? 2j : 8+2j) + 8*half
        FragBF afrag;
        #pragma unroll
        for (int j = 0; j < 8; ++j) {
            int kpos = ((j < 4) ? (2 * j) : (8 + 2 * j)) + 8 * half;
            afrag.u[j] = AsU[cur][l16 * 16 + (kpos >> 1)];
        }

        // B fragments: lane n = chunk*16 + l16; VGPR j holds K = (j<4 ? 2j : 8+2j) + 8*half.
        // Wc row n is contiguous in K, so read bf16 pairs straight from global.
        #pragma unroll
        for (int c = 0; c < 4; ++c) {
            int n = (wave * 4 + c) * 16 + l16;
            const unsigned int* Brow =
                WcB + (size_t)n * (CONV_DIM / 2) + it * 16;
            FragBF bfrag;
            #pragma unroll
            for (int j = 0; j < 8; ++j) bfrag.u[j] = Brow[((j < 4) ? j : (j + 4)) + 4 * half];

            acc[c] = __builtin_amdgcn_wmma_f32_16x16x32_bf16(
                false, afrag.v, false, bfrag.v, (short)0, acc[c], false, false);
        }

        __syncthreads();   // cur buffer fully read; next iter may overwrite it
    }

    // Epilogue: relu(acc + u[b, n]) * Wf[n], reduce over N
    // C/D layout: VGPR i, lane l -> row = i + 8*half, col n = chunk*16 + l16
    float tsum[8];
    #pragma unroll
    for (int i = 0; i < 8; ++i) tsum[i] = 0.0f;

    #pragma unroll
    for (int c = 0; c < 4; ++c) {
        int n = (wave * 4 + c) * 16 + l16;
        float wfv = Wf[n];
        #pragma unroll
        for (int i = 0; i < 8; ++i) {
            int b = bidx[i + 8 * half];
            float v = acc[c][i] + u[b * ATT_DIM + n];
            v = v > 0.0f ? v : 0.0f;
            tsum[i] += v * wfv;
        }
    }

    // cross-lane reduction within each 16-lane half, then LDS atomic combine
    #pragma unroll
    for (int i = 0; i < 8; ++i) {
        float s = tsum[i];
        #pragma unroll
        for (int m = 1; m < 16; m <<= 1) s += __shfl_xor(s, m, 32);
        if (l16 == 0) atomicAdd(&attp[i + 8 * half], s);
    }
    __syncthreads();
    if (t < 16) att[rowbase + t] = attp[t] + bfp[0];
}

// ---------------------------------------------------------------------------
// Pass 3: softmax over P=196 per batch
// ---------------------------------------------------------------------------
__global__ __launch_bounds__(256) void softmax_kernel(
    const float* __restrict__ att, float* __restrict__ alpha) {
    __shared__ float red[8];
    int b = blockIdx.x, t = threadIdx.x;
    int lane = t & 31, wave = t >> 5;
    float v = (t < NPIX) ? att[b * NPIX + t] : -3.0e38f;

    float m = v;
    #pragma unroll
    for (int s = 1; s < 32; s <<= 1) m = fmaxf(m, __shfl_xor(m, s, 32));
    if (lane == 0) red[wave] = m;
    __syncthreads();
    m = red[0];
    for (int i = 1; i < 8; ++i) m = fmaxf(m, red[i]);
    __syncthreads();

    float e = (t < NPIX) ? __expf(v - m) : 0.0f;
    float s = e;
    #pragma unroll
    for (int q = 1; q < 32; q <<= 1) s += __shfl_xor(s, q, 32);
    if (lane == 0) red[wave] = s;
    __syncthreads();
    float tot = 0.0f;
    for (int i = 0; i < 8; ++i) tot += red[i];
    if (t < NPIX) alpha[b * NPIX + t] = e / tot;
}

// ---------------------------------------------------------------------------
// Pass 4: out[b, c] = sum_p conv[b,p,c] * alpha[b,p]   (streaming, f32x4, NT)
// ---------------------------------------------------------------------------
__global__ __launch_bounds__(256) void weighted_sum_kernel(
    const float* __restrict__ conv, const float* __restrict__ alpha,
    float* __restrict__ out) {
    __shared__ float al[NPIX];
    int b = blockIdx.x, t = threadIdx.x;
    if (t < NPIX) al[t] = alpha[b * NPIX + t];
    __syncthreads();
    const f32x4* c4 = (const f32x4*)(conv + (size_t)b * NPIX * CONV_DIM);
    f32x4 a0 = {0.0f, 0.0f, 0.0f, 0.0f};
    f32x4 a1 = {0.0f, 0.0f, 0.0f, 0.0f};
    for (int p = 0; p < NPIX; ++p) {
        float w = al[p];
        f32x4 x = __builtin_nontemporal_load(c4 + (size_t)p * 512 + t);
        f32x4 y = __builtin_nontemporal_load(c4 + (size_t)p * 512 + 256 + t);
        a0 += x * w;
        a1 += y * w;
    }
    f32x4* o4 = (f32x4*)(out + (size_t)b * CONV_DIM);
    __builtin_nontemporal_store(a0, o4 + t);
    __builtin_nontemporal_store(a1, o4 + 256 + t);
}

// ---------------------------------------------------------------------------
extern "C" void kernel_launch(void* const* d_in, const int* in_sizes, int n_in,
                              void* d_out, int out_size, void* d_ws, size_t ws_size,
                              hipStream_t stream) {
    const float* conv = (const float*)d_in[0];
    const float* lstm = (const float*)d_in[1];
    const float* Wc   = (const float*)d_in[2];
    const float* bc   = (const float*)d_in[3];
    const float* Wl   = (const float*)d_in[4];
    const float* bl   = (const float*)d_in[5];
    const float* Wf   = (const float*)d_in[6];
    const float* bfp  = (const float*)d_in[7];
    float* out = (float*)d_out;

    char* ws = (char*)d_ws;
    unsigned int* WcB = (unsigned int*)ws;                              // 2 MB
    float* u     = (float*)(ws + (size_t)ATT_DIM * CONV_DIM * 2);       // 512 KB
    float* att   = u + BATCH * ATT_DIM;                                 // 200 KB
    float* alpha = att + NROWS;                                         // 200 KB

    wc_to_bf16_kernel<<<(ATT_DIM * CONV_DIM / 2) / 256, 256, 0, stream>>>(Wc, WcB);
    att2_kernel<<<BATCH, 256, 0, stream>>>(lstm, Wl, bc, bl, u);
    att_logits_kernel<<<NROWS / 16, 256, 0, stream>>>(conv, WcB, u, Wf, bfp, att);
    softmax_kernel<<<BATCH, 256, 0, stream>>>(att, alpha);
    weighted_sum_kernel<<<BATCH, 256, 0, stream>>>(conv, alpha, out);
}
